// QuantizedLinear_24721831756574
// MI455X (gfx1250) — matrix operand order repaired and checked by reading, not compiled, not run on hardware
//
#include <hip/hip_runtime.h>

typedef __attribute__((ext_vector_type(16))) _Float16 v16h;
typedef __attribute__((ext_vector_type(8)))  _Float16 v8h;
typedef __attribute__((ext_vector_type(4)))  _Float16 v4h;
typedef __attribute__((ext_vector_type(2)))  _Float16 v2h;
typedef __attribute__((ext_vector_type(8)))  float    v8f;
typedef __attribute__((ext_vector_type(4)))  int      v4i;

#define KDIM 4096
#define NDIM 4096

// ---------------------------------------------------------------------------
// Prep kernel 1: x (f32) -> xh (f16), 8 elements per thread.
// ---------------------------------------------------------------------------
__global__ __launch_bounds__(256)
void cvt_x_f16_kernel(const float* __restrict__ x, _Float16* __restrict__ xh, int n8)
{
    int i = blockIdx.x * 256 + threadIdx.x;
    if (i >= n8) return;
    size_t base = (size_t)i * 8;
    const float4 a = *(const float4*)&x[base];
    const float4 b = *(const float4*)&x[base + 4];
    v8h h = { (_Float16)a.x, (_Float16)a.y, (_Float16)a.z, (_Float16)a.w,
              (_Float16)b.x, (_Float16)b.y, (_Float16)b.z, (_Float16)b.w };
    *(v8h*)&xh[base] = h;
}

// ---------------------------------------------------------------------------
// Prep kernel 2: int4 W -> f16 W, one quant-group (8 weights) at a time.
// Full f32 dequant precision; runs exactly once per weight.
// ---------------------------------------------------------------------------
__global__ __launch_bounds__(256)
void dequant_w_f16_kernel(const int*   __restrict__ packed,
                          const float* __restrict__ scales,
                          const float* __restrict__ offsets,
                          _Float16*    __restrict__ wh, int ngroups)
{
    int g0 = (blockIdx.x * 256 + threadIdx.x) * 2;
    #pragma unroll
    for (int t = 0; t < 2; ++t) {
        int g = g0 + t;
        if (g >= ngroups) return;
        size_t flat = (size_t)g * 8;
        v4i  p = *(const v4i*)&packed[flat >> 1];
        float s = scales[g];
        float o = offsets[g];
        v8h w;
        #pragma unroll
        for (int j = 0; j < 4; ++j) {
            int b = p[j];
            w[2 * j]     = (_Float16)((float)(b & 15)        * s + o);
            w[2 * j + 1] = (_Float16)((float)((b >> 4) & 15) * s + o);
        }
        *(v8h*)&wh[flat] = w;
    }
}

// ---------------------------------------------------------------------------
// Main GEMM: out = xh @ wh^T + bias, pure f16 WMMA.
// Block tile 128(M) x 256(N), BK=64. 8 waves in 2(M) x 4(N); each wave owns a
// 64x64 sub-tile -> 16 f32 accumulators, 32 WMMAs per K-tile at a 1:1
// ds_load_b128 : wmma ratio. LDS double-buffered (96 KB of the 320 KB WGP
// pool, CDNA5-only). Tiles staged with GLOBAL_LOAD_ASYNC_TO_LDS_B128
// (ASYNCcnt) overlapped with compute; one barrier per K-tile.
// ---------------------------------------------------------------------------
#define BM 128
#define BN 256
#define BK 64

__global__ __launch_bounds__(256)
void gemm_f16_wmma_kernel(const _Float16* __restrict__ Xh,
                          const _Float16* __restrict__ Wh,
                          const float*    __restrict__ bias,
                          float*          __restrict__ out)
{
    __shared__ _Float16 As[2][BM][BK];   // 2 x 16 KB
    __shared__ _Float16 Bs[2][BN][BK];   // 2 x 32 KB

    const int tid  = threadIdx.x;
    const int lane = tid & 31;
    const int wave = tid >> 5;
    const int waveM = wave & 1;          // 0..1 -> 64-row M slab
    const int waveN = wave >> 1;         // 0..3 -> 64-col N slab
    const int blockN = blockIdx.x * BN;
    const int blockM = blockIdx.y * BM;

    // WMMA per-lane fragment coordinates (contiguous K order, same for A and B)
    const int aRow = lane & 15;
    const int aKb  = (lane >> 4) * 16;   // A: 16 contiguous K per lane
    const int bCol = lane & 15;
    const int bKb  = (lane >> 4) * 16;   // B: 16 contiguous K per lane

    v8f acc[4][4];
    #pragma unroll
    for (int i = 0; i < 4; ++i)
        #pragma unroll
        for (int j = 0; j < 4; ++j)
            acc[i][j] = (v8f)(0.0f);

    // Stage a 128x64 X tile (4 slots/thread) and 256x64 W tile (8 slots/thread)
    // into LDS buffer `buf` via async DMA. 16-byte granules per lane.
    auto stage = [&](int buf, int k0) {
        #pragma unroll
        for (int i = 0; i < 4; ++i) {
            int slot = tid + i * 256;
            int row  = slot >> 3;
            int col  = (slot & 7) * 8;   // in halves; 8 halves = 16 B
            unsigned ldsA = (unsigned)(size_t)&As[buf][row][col];
            unsigned long long gA =
                (unsigned long long)(size_t)&Xh[(size_t)(blockM + row) * KDIM + k0 + col];
            asm volatile("global_load_async_to_lds_b128 %0, %1, off"
                         :: "v"(ldsA), "v"(gA) : "memory");
        }
        #pragma unroll
        for (int i = 0; i < 8; ++i) {
            int slot = tid + i * 256;
            int row  = slot >> 3;
            int col  = (slot & 7) * 8;
            unsigned ldsB = (unsigned)(size_t)&Bs[buf][row][col];
            unsigned long long gB =
                (unsigned long long)(size_t)&Wh[(size_t)(blockN + row) * KDIM + k0 + col];
            asm volatile("global_load_async_to_lds_b128 %0, %1, off"
                         :: "v"(ldsB), "v"(gB) : "memory");
        }
    };

    int buf = 0;
    stage(0, 0);
    asm volatile("s_wait_asynccnt 0x0" ::: "memory");
    __syncthreads();

    for (int k0 = 0; k0 < KDIM; k0 += BK) {
        if (k0 + BK < KDIM)
            stage(buf ^ 1, k0 + BK);     // DMA next tile while we compute

        #pragma unroll
        for (int kk = 0; kk < 2; ++kk) {         // two 32-wide WMMA K-steps
            const int kw = kk * 32;
            // Hoist all 4 B fragments for this K-step (8 ds_load_b128)
            v16h bF[4];
            #pragma unroll
            for (int nf = 0; nf < 4; ++nf) {
                int n = waveN * 64 + nf * 16 + bCol;
                bF[nf] = *(const v16h*)&Bs[buf][n][kw + bKb];
            }
            #pragma unroll
            for (int mf = 0; mf < 4; ++mf) {
                int m = waveM * 64 + mf * 16 + aRow;
                v8h lo = *(const v8h*)&As[buf][m][kw + aKb];
                v8h hi = *(const v8h*)&As[buf][m][kw + aKb + 8];
                v16h aF;
                #pragma unroll
                for (int j = 0; j < 8; ++j) { aF[j] = lo[j]; aF[j + 8] = hi[j]; }
                #pragma unroll
                for (int nf = 0; nf < 4; ++nf) {
                    acc[mf][nf] = __builtin_amdgcn_wmma_f32_16x16x32_f16(
                        false, aF, false, bF[nf], (short)0, acc[mf][nf], false, false);
                }
            }
        }

        asm volatile("s_wait_asynccnt 0x0" ::: "memory");
        __syncthreads();
        buf ^= 1;
    }

    // Epilogue: C/D layout (VGPR r: M = r + 8*(lane/16), N = lane%16) + bias
    const int rowHi = (lane >> 4) * 8;
    #pragma unroll
    for (int mf = 0; mf < 4; ++mf) {
        #pragma unroll
        for (int nf = 0; nf < 4; ++nf) {
            int gn = blockN + waveN * 64 + nf * 16 + (lane & 15);
            float bv = bias[gn];
            #pragma unroll
            for (int r = 0; r < 8; ++r) {
                int gm = blockM + waveM * 64 + mf * 16 + rowHi + r;
                out[(size_t)gm * NDIM + gn] = acc[mf][nf][r] + bv;
            }
        }
    }
}

// ---------------------------------------------------------------------------
// Fallback fused kernel (used only if workspace is too small): single-pass
// design with magic-number (0x6400) nibble->f16 dequant + packed f16 FMA.
// ---------------------------------------------------------------------------
#define TILE_M 128
#define TILE_N 128
#define TILE_K 32

__global__ __launch_bounds__(256)
void qlinear_fused_kernel(const float* __restrict__ x,
                          const int*   __restrict__ packed,
                          const float* __restrict__ scales,
                          const float* __restrict__ offsets,
                          const float* __restrict__ bias,
                          float*       __restrict__ out)
{
    __shared__ _Float16 Xs[TILE_M][TILE_K];
    __shared__ _Float16 Ws[TILE_N][TILE_K];

    const int tid  = threadIdx.x;
    const int lane = tid & 31;
    const int wave = tid >> 5;
    const int waveM = wave & 3;
    const int waveN = wave >> 2;
    const int blockN = blockIdx.x * TILE_N;
    const int blockM = blockIdx.y * TILE_M;

    const int aRow = lane & 15;
    const int aKb  = (lane >> 4) * 16;
    const int bCol = lane & 15;
    const int bKb  = (lane >> 4) * 16;

    v8f acc[2][4];
    #pragma unroll
    for (int i = 0; i < 2; ++i)
        #pragma unroll
        for (int j = 0; j < 4; ++j)
            acc[i][j] = (v8f)(0.0f);

    const v2h kBias = { (_Float16)1024.0f, (_Float16)1024.0f };

    for (int k0 = 0; k0 < KDIM; k0 += TILE_K) {
        #pragma unroll
        for (int i = 0; i < 4; ++i) {
            int f   = tid + i * 256;
            int row = f >> 3;
            int c4  = (f & 7) * 4;
            const float4 v = *(const float4*)&x[(size_t)(blockM + row) * KDIM + (k0 + c4)];
            v4h h = { (_Float16)v.x, (_Float16)v.y, (_Float16)v.z, (_Float16)v.w };
            *(v4h*)&Xs[row][c4] = h;
        }
        #pragma unroll
        for (int i = 0; i < 2; ++i) {
            int g    = tid * 2 + i;
            int nLoc = g >> 2;
            int kLoc = (g & 3) * 8;
            size_t flat = (size_t)(blockN + nLoc) * KDIM + (size_t)(k0 + kLoc);
            v4i  p = *(const v4i*)&packed[flat >> 1];
            _Float16 sh = (_Float16)scales[flat >> 3];
            _Float16 oh = (_Float16)offsets[flat >> 3];
            v2h sv = { sh, sh };
            v2h ov = { oh, oh };
            v8h w;
            #pragma unroll
            for (int j = 0; j < 4; ++j) {
                unsigned b = (unsigned)p[j];
                unsigned t = (b & 15u) | ((b << 12) & 0x000F0000u) | 0x64006400u;
                v2h q;
                __builtin_memcpy(&q, &t, 4);     // {1024+lo, 1024+hi} exactly
                v2h wv = (q - kBias) * sv + ov;  // v_pk_add + v_pk_fma_f16
                w[2 * j] = wv[0]; w[2 * j + 1] = wv[1];
            }
            *(v8h*)&Ws[nLoc][kLoc] = w;
        }
        __syncthreads();

        v16h aF[2];
        #pragma unroll
        for (int mf = 0; mf < 2; ++mf) {
            int m = waveM * 32 + mf * 16 + aRow;
            v8h lo = *(const v8h*)&Xs[m][aKb];
            v8h hi = *(const v8h*)&Xs[m][aKb + 8];
            #pragma unroll
            for (int j = 0; j < 8; ++j) { aF[mf][j] = lo[j]; aF[mf][j + 8] = hi[j]; }
        }
        #pragma unroll
        for (int nf = 0; nf < 4; ++nf) {
            int n = waveN * 64 + nf * 16 + bCol;
            v16h bF = *(const v16h*)&Ws[n][bKb];
            #pragma unroll
            for (int mf = 0; mf < 2; ++mf) {
                acc[mf][nf] = __builtin_amdgcn_wmma_f32_16x16x32_f16(
                    false, aF[mf], false, bF, (short)0, acc[mf][nf], false, false);
            }
        }
        __syncthreads();
    }

    const int rowHi = (lane >> 4) * 8;
    #pragma unroll
    for (int mf = 0; mf < 2; ++mf) {
        #pragma unroll
        for (int nf = 0; nf < 4; ++nf) {
            int gn = blockN + waveN * 64 + nf * 16 + (lane & 15);
            float bv = bias[gn];
            #pragma unroll
            for (int r = 0; r < 8; ++r) {
                int gm = blockM + waveM * 32 + mf * 16 + rowHi + r;
                out[(size_t)gm * NDIM + gn] = acc[mf][nf][r] + bv;
            }
        }
    }
}

// ---------------------------------------------------------------------------
extern "C" void kernel_launch(void* const* d_in, const int* in_sizes, int n_in,
                              void* d_out, int out_size, void* d_ws, size_t ws_size,
                              hipStream_t stream) {
    const float* x       = (const float*)d_in[0];
    const int*   packed  = (const int*)d_in[1];
    const float* scales  = (const float*)d_in[2];
    const float* offsets = (const float*)d_in[3];
    const float* bias    = (const float*)d_in[4];
    float*       out     = (float*)d_out;

    const int    M        = in_sizes[0] / KDIM;          // 2048
    const size_t x_elems  = (size_t)in_sizes[0];         // M*K
    const size_t w_elems  = (size_t)NDIM * KDIM;         // N*K
    const size_t need     = (x_elems + w_elems) * sizeof(_Float16);  // ~48 MB

    if (ws_size >= need) {
        _Float16* xh = (_Float16*)d_ws;
        _Float16* wh = (_Float16*)((char*)d_ws + x_elems * sizeof(_Float16));

        const int n8 = (int)(x_elems / 8);
        cvt_x_f16_kernel<<<(n8 + 255) / 256, 256, 0, stream>>>(x, xh, n8);

        const int ngroups = (int)(w_elems / 8);
        const int nthr    = (ngroups + 1) / 2;
        dequant_w_f16_kernel<<<(nthr + 255) / 256, 256, 0, stream>>>(
            packed, scales, offsets, wh, ngroups);

        dim3 grid(NDIM / BN, M / BM);                    // 16 x 16
        gemm_f16_wmma_kernel<<<grid, 256, 0, stream>>>(xh, wh, bias, out);
    } else {
        dim3 grid(NDIM / TILE_N, M / TILE_M);
        qlinear_fused_kernel<<<grid, 256, 0, stream>>>(x, packed, scales, offsets, bias, out);
    }
}
